// GraphEncoder_43533788512744
// MI455X (gfx1250) — matrix operand order repaired and checked by reading, not compiled, not run on hardware
//
#include <hip/hip_runtime.h>
#include <hip/hip_bf16.h>
#include <math.h>

// ---------------------------------------------------------------------------
// GraphEncoder pipeline for MI455X (gfx1250, wave32, WMMA).
// GEMMs: v_wmma_f32_16x16x32_bf16 with async-to-LDS double-buffered B panels.
// ---------------------------------------------------------------------------

#define BB   32
#define LL   256
#define NN   (BB * LL)      // 8192
#define EE   512
#define HH   512
#define HC   256
#define G4   1024           // 4*Hc
#define DEG  20
#define SS   10
#define NLAYERS 7
#define LDB_S 40            // LDS row stride (ushorts): 80B, 16B-aligned, bank-spread

#if __has_builtin(__builtin_amdgcn_global_load_async_to_lds_b128) && \
    __has_builtin(__builtin_amdgcn_s_wait_asynccnt)
#define USE_ASYNC_LDS 1
#else
#define USE_ASYNC_LDS 0
#endif

typedef __attribute__((ext_vector_type(16))) __bf16        bf16x16;
typedef __attribute__((ext_vector_type(8)))  unsigned short ushort8;
typedef __attribute__((ext_vector_type(8)))  float          floatx8;

#if USE_ASYNC_LDS
// exact parameter types of __builtin_amdgcn_global_load_async_to_lds_b128:
// (v4i addrspace(1)*, v4i addrspace(3)*, imm i32 offset, imm i32 cpol)
typedef int v4i_vs __attribute__((vector_size(16)));
typedef __attribute__((address_space(1))) v4i_vs* as1_v4i;
typedef __attribute__((address_space(3))) v4i_vs* as3_v4i;
#endif

union FragU { bf16x16 bf; ushort8 u[2]; };

// NOTE: placeholders for jax.random.permutation(key(42/43), 20)[:10]
// (not reproducible in HIP; compile-only loop, no correctness check).
__device__ __constant__ int PF10[SS] = {3, 17, 8, 0, 12, 19, 5, 14, 9, 1};
__device__ __constant__ int PB10[SS] = {11, 2, 18, 7, 15, 4, 10, 16, 6, 13};

// ---- bf16 helpers (storage type = unsigned short, RNE rounding) -----------
static __device__ __forceinline__ unsigned short f32_to_bf16(float f) {
  unsigned int u = __float_as_uint(f);
  u += 0x7FFFu + ((u >> 16) & 1u);          // round-to-nearest-even
  return (unsigned short)(u >> 16);
}
static __device__ __forceinline__ float bf16_to_f32(unsigned short s) {
  return __uint_as_float(((unsigned int)s) << 16);
}

// ---- WMMA fragment loaders (CDNA5 16-bit layouts, wave32) -----------------
// A (16xK tile, row-major, K-contiguous): lane = M; lanes 0-15 -> K base 0,
// lanes 16-31 -> K base 8; elems 0..7 = K kb+0..7, elems 8..15 = K kb+16..23.
static __device__ __forceinline__ bf16x16
load_frag_a(const unsigned short* __restrict__ tile, int lda, int lane) {
  int m  = lane & 15;
  int kb = (lane >> 4) << 3;                 // 0 or 8
  const unsigned short* p = tile + (size_t)m * lda + kb;
  FragU f;
  f.u[0] = *(const ushort8*)(p);
  f.u[1] = *(const ushort8*)(p + 16);
  return f.bf;
}
// B (KxN tile) built from row-major Bm[N, K] (i.e. B^T, K-contiguous rows):
// lane = N; lanes 0-15 -> K base 0, lanes 16-31 -> K base 8; elems 8..15 = K kb+16..23.
static __device__ __forceinline__ bf16x16
load_frag_b(const unsigned short* __restrict__ tile, int ldb, int lane) {
  int n  = lane & 15;
  int kb = (lane >> 4) << 3;                 // 0 or 8
  const unsigned short* p = tile + (size_t)n * ldb + kb;
  FragU f;
  f.u[0] = *(const ushort8*)(p);
  f.u[1] = *(const ushort8*)(p + 16);
  return f.bf;
}

static __device__ __forceinline__ floatx8
wmma_bf16(bf16x16 a, bf16x16 b, floatx8 c) {
  return __builtin_amdgcn_wmma_f32_16x16x32_bf16(
      false, a, false, b, (short)0, c, false, false);
}

static __device__ __forceinline__ float sigf(float x) {
  return 1.0f / (1.0f + __expf(-x));
}

// ---- async B-panel staging: 64 rows x 32 bf16 -> LDS (stride LDB_S) -------
// 256 threads, each moves 16 bytes: one global_load_async_to_lds_b128 / wave.
static __device__ __forceinline__ void
copy_b_panel(unsigned short* dstLds, const unsigned short* srcG,
             int ldb, int tid) {
  int r   = tid >> 2;            // 0..63 (panel row)
  int seg = (tid & 3) << 3;      // 0,8,16,24 (8 ushorts = 16B)
  const unsigned short* g = srcG + (size_t)r * ldb + seg;
  unsigned short* l = dstLds + r * LDB_S + seg;
#if USE_ASYNC_LDS
  __builtin_amdgcn_global_load_async_to_lds_b128(
      (as1_v4i)(void*)g, (as3_v4i)(void*)l, 0, 0);
#else
  *(ushort8*)l = *(const ushort8*)g;
#endif
}
static __device__ __forceinline__ void wait_async() {
#if USE_ASYNC_LDS
  __builtin_amdgcn_s_wait_asynccnt(0);
#endif
}

// ---------------------------------------------------------------------------
// Generic bf16 WMMA GEMM: C[M,N] = concat_K(A0,A1)[M,K] * B[N,K]^T + bias
// Block: 256 thr (8 waves). Block tile: 128(M) x 64(N); wave tile 16 x 64.
// B panel (shared by all 8 waves) is double-buffered in LDS via async copy.
// Writes f32 C (ldc) and optionally bf16 copy (Cb, ldcb). relu optional.
// ---------------------------------------------------------------------------
__global__ __launch_bounds__(256)
void gemm_bf16_kernel(const unsigned short* __restrict__ A0,
                      const unsigned short* __restrict__ A1, int lda, int Ksplit,
                      const unsigned short* __restrict__ Bm, int ldb,
                      const float* __restrict__ bias,
                      float* __restrict__ Cf, int ldc,
                      unsigned short* __restrict__ Cb, int ldcb,
                      int K, int relu) {
  __shared__ unsigned short Bs[2][64 * LDB_S];   // 2 x 5 KB

  const int tid   = threadIdx.x;
  const int lane  = tid & 31;
  const int wv    = tid >> 5;
  const int mBase = blockIdx.x * 128 + wv * 16;
  const int nBase = blockIdx.y * 64;

  floatx8 acc[4];
#pragma unroll
  for (int s = 0; s < 4; ++s) acc[s] = (floatx8)0.0f;

  // prologue: stage panel for k0 = 0
  copy_b_panel(Bs[0], Bm + (size_t)nBase * ldb, ldb, tid);
  wait_async();
  __syncthreads();

  int cur = 0;
  for (int k0 = 0; k0 < K; k0 += 32) {
    if (k0 + 32 < K)                                    // stage next panel
      copy_b_panel(Bs[cur ^ 1], Bm + (size_t)nBase * ldb + k0 + 32, ldb, tid);

    const unsigned short* As;
    int kk;
    if (k0 < Ksplit) { As = A0; kk = k0; }
    else             { As = A1; kk = k0 - Ksplit; }
    bf16x16 a = load_frag_a(As + (size_t)mBase * lda + kk, lda, lane);
#pragma unroll
    for (int s = 0; s < 4; ++s) {
      bf16x16 b = load_frag_b(&Bs[cur][(16 * s) * LDB_S], LDB_S, lane);
      acc[s] = wmma_bf16(a, b, acc[s]);
    }

    wait_async();        // this wave's async copy of the next panel landed
    __syncthreads();     // all waves' copies landed; safe to flip buffers
    cur ^= 1;
  }

  const int nCol = lane & 15;
  const int rOff = (lane >> 4) << 3;         // 0 or 8
#pragma unroll
  for (int s = 0; s < 4; ++s) {
    int col  = nBase + 16 * s + nCol;
    float bv = bias ? bias[col] : 0.0f;
#pragma unroll
    for (int v = 0; v < 8; ++v) {
      int row = mBase + rOff + v;
      float x = acc[s][v] + bv;
      if (relu) x = fmaxf(x, 0.0f);
      Cf[(size_t)row * ldc + col] = x;
      if (Cb) Cb[(size_t)row * ldcb + col] = f32_to_bf16(x);
    }
  }
}

// ---------------------------------------------------------------------------
// Embedding gather -> bf16 features [N, E]
// ---------------------------------------------------------------------------
__global__ __launch_bounds__(256)
void embed_kernel(const int* __restrict__ feat_ids,
                  const float* __restrict__ emb,
                  unsigned short* __restrict__ featb) {
  size_t i = (size_t)blockIdx.x * blockDim.x + threadIdx.x;   // N*E = 4M
  int n = (int)(i >> 9);
  int e = (int)(i & 511);
  int tok = feat_ids[n];
  featb[i] = f32_to_bf16(emb[(size_t)tok * EE + e]);
}

// ---------------------------------------------------------------------------
// Weight prep: f32 -> bf16 copy, and layer-wise transpose for SAGE weights
// ---------------------------------------------------------------------------
__global__ __launch_bounds__(256)
void cvt_bf16_kernel(const float* __restrict__ s, unsigned short* __restrict__ d,
                     int n) {
  int i = blockIdx.x * blockDim.x + threadIdx.x;
  if (i < n) d[i] = f32_to_bf16(s[i]);
}

// W: [layers][1024][512] (row=k, col=h) -> Wt: [layers][512][1024] (row=h, K-major)
__global__ __launch_bounds__(256)
void transpose_w_kernel(const float* __restrict__ W,
                        unsigned short* __restrict__ Wt) {
  size_t i = (size_t)blockIdx.x * blockDim.x + threadIdx.x;
  size_t total = (size_t)NLAYERS * 1024 * 512;
  if (i >= total) return;
  int l   = (int)(i / (1024 * 512));
  int rem = (int)(i % (1024 * 512));
  int r = rem / 512, c = rem % 512;
  Wt[(size_t)l * 512 * 1024 + (size_t)c * 1024 + r] = f32_to_bf16(W[i]);
}

// ---------------------------------------------------------------------------
// BiLSTM recurrence. grid.x = 2 (dir), block = 1024 thr = 32 waves.
// Wave (mt,jt) owns gate tile rows [mt*16,+16), cols [jt*16,+16) of [32,1024];
// cell state lives in accumulator-layout registers; h (bf16) lives in LDS.
// ---------------------------------------------------------------------------
__global__ __launch_bounds__(1024)
void lstm_kernel(const float* __restrict__ xgF, const float* __restrict__ xgB,
                 const unsigned short* __restrict__ whhF,
                 const unsigned short* __restrict__ whhB,
                 float* __restrict__ outVec) {
  __shared__ unsigned short hS[BB * HC];      // 16 KB, bf16 hidden state

  const int dir  = blockIdx.x;
  const float* xg = dir ? xgB : xgF;
  const unsigned short* whh = dir ? whhB : whhF;

  const int tid  = threadIdx.x;
  const int lane = tid & 31;
  const int wv   = tid >> 5;                  // 0..31
  const int mt   = wv >> 4;                   // 0..1   (16-row half of batch)
  const int jt   = wv & 15;                   // 0..15  (16-col tile of Hc)

  for (int i = tid; i < BB * HC; i += 1024) hS[i] = 0;
  floatx8 cReg = (floatx8)0.0f;
  __syncthreads();

  const int nCol = lane & 15;
  const int rOff = (lane >> 4) << 3;          // 0 or 8
  const int col  = jt * 16 + nCol;            // hidden unit j

  for (int t = 0; t < LL; ++t) {
    const int tt = dir ? (LL - 1 - t) : t;    // natural-time index

    floatx8 g[4];
#pragma unroll
    for (int q = 0; q < 4; ++q) g[q] = (floatx8)0.0f;

#pragma unroll
    for (int k0 = 0; k0 < HC; k0 += 32) {
      bf16x16 a = load_frag_a(hS + (size_t)(mt * 16) * HC + k0, HC, lane);
#pragma unroll
      for (int q = 0; q < 4; ++q) {
        const unsigned short* brow =
            whh + (size_t)(q * HC + jt * 16) * HC + k0;
        g[q] = wmma_bf16(a, load_frag_b(brow, HC, lane), g[q]);
      }
    }

    __syncthreads();                          // all hS reads done (WAR)

#pragma unroll
    for (int v = 0; v < 8; ++v) {
      int b = mt * 16 + rOff + v;             // batch row
      const float* xr = xg + ((size_t)b * LL + tt) * G4;
      float iv = sigf (g[0][v] + xr[          col]);
      float fv = sigf (g[1][v] + xr[    HC + col]);
      float gv = tanhf(g[2][v] + xr[2 * HC + col]);
      float ov = sigf (g[3][v] + xr[3 * HC + col]);
      float c  = fv * cReg[v] + iv * gv;
      cReg[v]  = c;
      float h  = ov * tanhf(c);
      hS[(size_t)b * HC + col] = f32_to_bf16(h);
      outVec[((size_t)b * LL + tt) * HH + dir * HC + col] = h;
    }
    __syncthreads();                          // hS writes visible (RAW)
  }
}

// ---------------------------------------------------------------------------
// Neighbor gather + mean. grid (N, 2), block 512 (one thread per column).
// layer0: table = node_repres (f32), also seeds bf16 self buffer.
// layer>0: table = selfRead (bf16 output of previous layer).
// ---------------------------------------------------------------------------
__global__ __launch_bounds__(512)
void gather_mean_kernel(const int* __restrict__ adjF, const int* __restrict__ adjB,
                        const float* __restrict__ nodeRep,
                        const unsigned short* __restrict__ selfF,
                        const unsigned short* __restrict__ selfB,
                        unsigned short* __restrict__ meanF,
                        unsigned short* __restrict__ meanB,
                        unsigned short* __restrict__ seedF,
                        unsigned short* __restrict__ seedB,
                        int layer0) {
  const int n   = blockIdx.x;
  const int dir = blockIdx.y;
  const int col = threadIdx.x;
  const int* adj = dir ? adjB : adjF;
  const unsigned short* selfT = dir ? selfB : selfF;
  unsigned short* mean = dir ? meanB : meanF;
  const int* perm = dir ? PB10 : PF10;

  float s = 0.0f;
#pragma unroll
  for (int k = 0; k < SS; ++k) {
    int m = adj[(size_t)n * DEG + perm[k]];
    float v = layer0 ? nodeRep[(size_t)m * HH + col]
                     : bf16_to_f32(selfT[(size_t)m * HH + col]);
    s += v;
  }
  mean[(size_t)n * HH + col] = f32_to_bf16(s * (1.0f / SS));
  if (layer0) {
    unsigned short* seed = dir ? seedB : seedF;
    seed[(size_t)n * HH + col] = f32_to_bf16(nodeRep[(size_t)n * HH + col]);
  }
}

// ---------------------------------------------------------------------------
// Max-pool over L and pack outputs. grid (32, 4), block 256 -> 1024 cols.
// ---------------------------------------------------------------------------
__global__ __launch_bounds__(256)
void pool_pack_kernel(const float* __restrict__ hidF,
                      const float* __restrict__ hidB,
                      float* __restrict__ outHidden,
                      float* __restrict__ outPooled) {
  const int b   = blockIdx.x;
  const int col = blockIdx.y * blockDim.x + threadIdx.x;     // 0..1023
  float mx = -3.4e38f;
  for (int l = 0; l < LL; ++l) {
    size_t n = (size_t)b * LL + l;
    float v = (col < HH) ? hidF[n * HH + col] : hidB[n * HH + (col - HH)];
    outHidden[n * (2 * HH) + col] = v;
    mx = fmaxf(mx, v);
  }
  outPooled[(size_t)b * (2 * HH) + col] = mx;
}

// ---------------------------------------------------------------------------
// Host-side orchestration
// ---------------------------------------------------------------------------
static inline size_t alignUp(size_t x) { return (x + 255) & ~(size_t)255; }

extern "C" void kernel_launch(void* const* d_in, const int* in_sizes, int n_in,
                              void* d_out, int out_size, void* d_ws, size_t ws_size,
                              hipStream_t stream) {
  (void)in_sizes; (void)n_in; (void)out_size; (void)ws_size;

  const int*   fw_adj = (const int*)  d_in[0];
  const int*   bw_adj = (const int*)  d_in[1];
  const int*   featId = (const int*)  d_in[2];
  const float* emb    = (const float*)d_in[4];
  const float* WihF   = (const float*)d_in[5];
  const float* WhhF   = (const float*)d_in[6];
  const float* bF     = (const float*)d_in[7];
  const float* WihB   = (const float*)d_in[8];
  const float* WhhB   = (const float*)d_in[9];
  const float* bB     = (const float*)d_in[10];
  const float* fwW    = (const float*)d_in[11];
  const float* fwb    = (const float*)d_in[12];
  const float* bwW    = (const float*)d_in[13];
  const float* bwb    = (const float*)d_in[14];

  // d_out layout: hidden [32,256,1024] | pooled [32,1024] | output_vector [8192,512]
  float* outHidden = (float*)d_out;
  float* outPooled = outHidden + (size_t)BB * LL * 2 * HH;
  float* outVec    = outPooled + (size_t)BB * 2 * HH;

  // workspace carve
  char* p = (char*)d_ws;
  auto carve = [&](size_t bytes) { void* r = (void*)p; p += alignUp(bytes); return r; };
  unsigned short* wihF_bf = (unsigned short*)carve((size_t)G4 * EE * 2);
  unsigned short* wihB_bf = (unsigned short*)carve((size_t)G4 * EE * 2);
  unsigned short* whhF_bf = (unsigned short*)carve((size_t)G4 * HC * 2);
  unsigned short* whhB_bf = (unsigned short*)carve((size_t)G4 * HC * 2);
  unsigned short* WtF     = (unsigned short*)carve((size_t)NLAYERS * HH * 1024 * 2);
  unsigned short* WtB     = (unsigned short*)carve((size_t)NLAYERS * HH * 1024 * 2);
  unsigned short* featb   = (unsigned short*)carve((size_t)NN * EE * 2);
  float*          xgF     = (float*)carve((size_t)NN * G4 * 4);
  float*          xgB     = (float*)carve((size_t)NN * G4 * 4);
  unsigned short* selfF0  = (unsigned short*)carve((size_t)NN * HH * 2);
  unsigned short* selfF1  = (unsigned short*)carve((size_t)NN * HH * 2);
  unsigned short* selfB0  = (unsigned short*)carve((size_t)NN * HH * 2);
  unsigned short* selfB1  = (unsigned short*)carve((size_t)NN * HH * 2);
  unsigned short* meanF   = (unsigned short*)carve((size_t)NN * HH * 2);
  unsigned short* meanB   = (unsigned short*)carve((size_t)NN * HH * 2);
  float*          hidF    = (float*)carve((size_t)NN * HH * 4);
  float*          hidB    = (float*)carve((size_t)NN * HH * 4);

  // --- weight prep ---
  cvt_bf16_kernel<<<(G4 * EE + 255) / 256, 256, 0, stream>>>(WihF, wihF_bf, G4 * EE);
  cvt_bf16_kernel<<<(G4 * EE + 255) / 256, 256, 0, stream>>>(WihB, wihB_bf, G4 * EE);
  cvt_bf16_kernel<<<(G4 * HC + 255) / 256, 256, 0, stream>>>(WhhF, whhF_bf, G4 * HC);
  cvt_bf16_kernel<<<(G4 * HC + 255) / 256, 256, 0, stream>>>(WhhB, whhB_bf, G4 * HC);
  {
    size_t tot = (size_t)NLAYERS * 1024 * 512;
    transpose_w_kernel<<<(unsigned)((tot + 255) / 256), 256, 0, stream>>>(fwW, WtF);
    transpose_w_kernel<<<(unsigned)((tot + 255) / 256), 256, 0, stream>>>(bwW, WtB);
  }

  // --- embedding gather (bf16) ---
  embed_kernel<<<(NN * EE) / 256, 256, 0, stream>>>(featId, emb, featb);

  // --- input-gate GEMMs: xg = feat @ Wih^T + b   [8192,512]x[512,1024] ---
  gemm_bf16_kernel<<<dim3(NN / 128, G4 / 64), 256, 0, stream>>>(
      featb, featb, EE, EE, wihF_bf, EE, bF, xgF, G4, nullptr, 0, EE, 0);
  gemm_bf16_kernel<<<dim3(NN / 128, G4 / 64), 256, 0, stream>>>(
      featb, featb, EE, EE, wihB_bf, EE, bB, xgB, G4, nullptr, 0, EE, 0);

  // --- BiLSTM recurrence (writes output_vector / node_repres) ---
  lstm_kernel<<<2, 1024, 0, stream>>>(xgF, xgB, whhF_bf, whhB_bf, outVec);

  // --- GraphSAGE layers ---
  unsigned short* sF[2] = {selfF0, selfF1};
  unsigned short* sB[2] = {selfB0, selfB1};
  for (int l = 0; l < NLAYERS; ++l) {
    unsigned short* rF = sF[l & 1];
    unsigned short* wF = sF[(l + 1) & 1];
    unsigned short* rB = sB[l & 1];
    unsigned short* wB = sB[(l + 1) & 1];

    gather_mean_kernel<<<dim3(NN, 2), 512, 0, stream>>>(
        fw_adj, bw_adj, outVec, rF, rB, meanF, meanB, rF, rB, l == 0 ? 1 : 0);

    // fw: hid = relu([self|mean] @ W_l + b_l), bf16 copy -> next self
    gemm_bf16_kernel<<<dim3(NN / 128, HH / 64), 256, 0, stream>>>(
        rF, meanF, HH, HH, WtF + (size_t)l * HH * 1024, 1024,
        fwb + (size_t)l * HH, hidF, HH, wF, HH, 1024, 1);
    // bw
    gemm_bf16_kernel<<<dim3(NN / 128, HH / 64), 256, 0, stream>>>(
        rB, meanB, HH, HH, WtB + (size_t)l * HH * 1024, 1024,
        bwb + (size_t)l * HH, hidB, HH, wB, HH, 1024, 1);
  }

  // --- pool + pack ---
  pool_pack_kernel<<<dim3(BB, 4), 256, 0, stream>>>(hidF, hidB, outHidden, outPooled);
}
